// SimpleMLA_81106162418387
// MI455X (gfx1250) — matrix operand order repaired and checked by reading, not compiled, not run on hardware
//
#include <hip/hip_runtime.h>
#include <hip/hip_bf16.h>
#include <math.h>

// ---------------------------------------------------------------------------
// Types for CDNA5 WMMA (wave32): V_WMMA_F32_16X16X32_BF16
// ---------------------------------------------------------------------------
typedef __attribute__((ext_vector_type(16))) __bf16 v16bf;
typedef __attribute__((ext_vector_type(8)))  float  v8f;

union Frag16 {            // 16 bf16 = 32 bytes = 8 VGPRs
    v16bf v;
    uint4 u[2];
};

__device__ __forceinline__ unsigned short f32_to_bf16u(float f) {
    unsigned int u = __float_as_uint(f);
    u += 0x7FFFu + ((u >> 16) & 1u);   // round-to-nearest-even
    return (unsigned short)(u >> 16);
}

// ---------------------------------------------------------------------------
// Problem constants
// ---------------------------------------------------------------------------
#define C_B   2
#define C_S   1024
#define C_H   4096
#define C_NH  32
#define C_QR  1536
#define C_KVR 512
#define C_RD  64
#define C_ND  128
#define C_VD  128
#define C_QK  192          // RD + ND
#define C_M   (C_B * C_S)  // 2048 rows of activations
#define KVPAD 768          // padded kv width (multiple of 256)

// ---------------------------------------------------------------------------
// Tensor Data Mover support (guarded: falls back to manual staging)
// ---------------------------------------------------------------------------
#if defined(__gfx1250__) && __has_builtin(__builtin_amdgcn_tensor_load_to_lds)
#define HAVE_TDM 1
#endif

#ifdef HAVE_TDM
typedef unsigned int tdm_v4u __attribute__((ext_vector_type(4)));
typedef int          tdm_v8i __attribute__((ext_vector_type(8)));
typedef int          tdm_v4i __attribute__((ext_vector_type(4)));

__device__ __forceinline__ unsigned lds_offset_of(const void* p) {
    // flat->LDS addrspace cast truncates to the 32-bit LDS byte offset
    return (unsigned)(unsigned long long)
        (__attribute__((address_space(3))) const char*)(const char*)p;
}

// 2D bf16 tile load: tileW elems per row, tileH rows, row stride strideElems.
// Optional LDS padding: padIntervalCode per D# (3 => every 16 DWORDs),
// padAmountCode per D# (3 => 4 DWORDs = 8 bf16).
__device__ __forceinline__ void tdm_load_2d_bf16(
    unsigned ldsAddr, const void* gptr,
    unsigned tileW, unsigned tileH, unsigned strideElems,
    unsigned padIntervalCode, unsigned padAmountCode, bool padEnable)
{
    unsigned long long ga = (unsigned long long)(size_t)gptr;
    tdm_v4u g0 = {
        1u,                                              // count=1 (valid D#)
        ldsAddr,                                         // lds_addr [63:32]
        (unsigned)(ga & 0xffffffffull),                  // global_addr lo
        (unsigned)((ga >> 32) & 0x01ffffffull)           // global_addr hi
            | 0x80000000u                                // type=2 ("image")
    };
    const unsigned long long TD = 1ull << 20;            // generous tensor dims
    unsigned w0 = (1u << 16)                             // data_size = 2 bytes
                | (padEnable ? (1u << 20) : 0u)          // pad_enable
                | (padIntervalCode << 22)
                | (padAmountCode << 25);
    unsigned w1 = (unsigned)((TD & 0xffffull) << 16);    // tensor_dim0[15:0]
    unsigned w2 = (unsigned)((TD >> 16) & 0xffffull)     // tensor_dim0[31:16]
                | (unsigned)((TD & 0xffffull) << 16);    // tensor_dim1[15:0]
    unsigned w3 = (unsigned)((TD >> 16) & 0xffffull)     // tensor_dim1[31:16]
                | (tileW << 16);                         // tile_dim0
    unsigned w4 = tileH & 0xffffu;                       // tile_dim1 (dim2=0)
    unsigned w5 = strideElems;                           // tensor_dim0_stride lo
    unsigned w6 = 0u;                                    // stride hi / dim1_stride
    tdm_v8i g1 = { (int)w0, (int)w1, (int)w2, (int)w3,
                   (int)w4, (int)w5, (int)w6, 0 };
    tdm_v4i z4 = { 0, 0, 0, 0 };
#if __has_include(<hip/amd_detail/amd_gfx1250_TDM.h>)
    tdm_v8i z8 = { 0, 0, 0, 0, 0, 0, 0, 0 };
    __builtin_amdgcn_tensor_load_to_lds(g0, g1, z4, z4, z8, 0);
#else
    __builtin_amdgcn_tensor_load_to_lds(g0, g1, z4, z4, 0);
#endif
}

__device__ __forceinline__ void tdm_wait0() {
#if __has_builtin(__builtin_amdgcn_s_wait_tensorcnt)
    __builtin_amdgcn_s_wait_tensorcnt(0);
#endif
}
#endif // HAVE_TDM

// ---------------------------------------------------------------------------
// fp32 -> bf16 with optional column zero-padding (dst has dcols >= cols)
// ---------------------------------------------------------------------------
__global__ void cvt_f32_bf16_pad(const float* __restrict__ src,
                                 unsigned short* __restrict__ dst,
                                 int rows, int cols, int dcols) {
    size_t i = (size_t)blockIdx.x * blockDim.x + threadIdx.x;
    size_t total = (size_t)rows * dcols;
    if (i >= total) return;
    int r = (int)(i / dcols);
    int c = (int)(i % dcols);
    float v = (c < cols) ? src[(size_t)r * cols + c] : 0.0f;
    dst[i] = f32_to_bf16u(v);
}

// ---------------------------------------------------------------------------
// Tiled bf16 WMMA GEMM: C(f32, MxN) = A(bf16, MxK, lda) * B(bf16, KxN, ldb)
// Block tile 64x256, BK=32, 8 waves (2x4), each wave 32x64 (2x4 WMMA tiles).
// A tile staged via TDM (padded LDS stride), B tile transposed in LDS so
// fragments are contiguous ds_load_b128s matching the ISA VGPR layouts.
// ---------------------------------------------------------------------------
#define GBM 64
#define GBN 256
#define GBK 32
#define ASTR 40   // LDS row stride (elements): 32 data + 8 pad
#define BSTR 40

__global__ __launch_bounds__(256) void gemm_bf16_wmma(
    const unsigned short* __restrict__ A,
    const unsigned short* __restrict__ B,
    float* __restrict__ C,
    int K, int lda, int ldb, int ldc) {

    __shared__ alignas(16) unsigned short As[GBM * ASTR];
    __shared__ alignas(16) unsigned short Bs[GBN * BSTR];

    const int tid  = threadIdx.x;
    const int wave = tid >> 5;
    const int lane = tid & 31;
    const int half = lane >> 4;
    const int l16  = lane & 15;
    const int wm   = wave >> 2;   // 0..1 -> 32 rows each
    const int wn   = wave & 3;    // 0..3 -> 64 cols each
    const int blockRow = blockIdx.y * GBM;
    const int blockCol = blockIdx.x * GBN;

#ifdef HAVE_TDM
    const unsigned AsBase = lds_offset_of(&As[0]);
#endif

    v8f acc[2][4];
    const v8f vzero = {0.f, 0.f, 0.f, 0.f, 0.f, 0.f, 0.f, 0.f};
#pragma unroll
    for (int mt = 0; mt < 2; ++mt)
#pragma unroll
        for (int nt = 0; nt < 4; ++nt) acc[mt][nt] = vzero;

    for (int k0 = 0; k0 < K; k0 += GBK) {
        __syncthreads();
        // ---- stage A tile (64 x 32)
#ifdef HAVE_TDM
        if (wave == 0) {
            tdm_load_2d_bf16(AsBase, A + (size_t)blockRow * lda + k0,
                             GBK, GBM, (unsigned)lda,
                             /*interval 16 DW*/3u, /*pad 4 DW*/3u, true);
        }
#else
        {
            int r = tid >> 2;               // 0..63
            int c = (tid & 3) << 3;         // 0,8,16,24
            *(uint4*)&As[r * ASTR + c] =
                *(const uint4*)(A + (size_t)(blockRow + r) * lda + k0 + c);
        }
#endif
        // ---- stage B tile transposed (Bs[n][k]), 32 n's per thread
        {
            int kr = tid >> 3;              // 0..31
            int n0 = (tid & 7) << 5;        // 0..224
            const uint4* src = (const uint4*)(B + (size_t)(k0 + kr) * ldb + blockCol + n0);
            uint4 p0 = src[0];
            uint4 p1 = src[1];
            uint4 p2 = src[2];
            uint4 p3 = src[3];
            unsigned short tmp[32];
            *(uint4*)&tmp[0]  = p0;
            *(uint4*)&tmp[8]  = p1;
            *(uint4*)&tmp[16] = p2;
            *(uint4*)&tmp[24] = p3;
#pragma unroll
            for (int i = 0; i < 32; ++i) Bs[(n0 + i) * BSTR + kr] = tmp[i];
        }
#ifdef HAVE_TDM
        if (wave == 0) tdm_wait0();
#endif
        __syncthreads();

        // ---- fragments (ISA 16-bit A 16x32 / B 32x16 layouts)
        Frag16 afr[2], bfr[4];
#pragma unroll
        for (int mt = 0; mt < 2; ++mt) {
            int m = wm * 32 + mt * 16 + l16;
            afr[mt].u[0] = *(const uint4*)&As[m * ASTR + half * 8];
            afr[mt].u[1] = *(const uint4*)&As[m * ASTR + 16 + half * 8];
        }
#pragma unroll
        for (int nt = 0; nt < 4; ++nt) {
            int n = wn * 64 + nt * 16 + l16;
            bfr[nt].u[0] = *(const uint4*)&Bs[n * BSTR + half * 8];
            bfr[nt].u[1] = *(const uint4*)&Bs[n * BSTR + 16 + half * 8];
        }
#pragma unroll
        for (int mt = 0; mt < 2; ++mt)
#pragma unroll
            for (int nt = 0; nt < 4; ++nt)
                acc[mt][nt] = __builtin_amdgcn_wmma_f32_16x16x32_bf16(
                    false, afr[mt].v, false, bfr[nt].v,
                    (short)0, acc[mt][nt], false, false);
    }

    // ---- store C (f32). C-layout: row = half*8 + r, col = l16 within tile
#pragma unroll
    for (int mt = 0; mt < 2; ++mt)
#pragma unroll
        for (int nt = 0; nt < 4; ++nt)
#pragma unroll
            for (int r = 0; r < 8; ++r) {
                int row = blockRow + wm * 32 + mt * 16 + half * 8 + r;
                int col = blockCol + wn * 64 + nt * 16 + l16;
                C[(size_t)row * ldc + col] = acc[mt][nt][r];
            }
}

// ---------------------------------------------------------------------------
// RMSNorm per row: y = x / (sqrt(sum x^2)*C^-0.5 + eps) * g, output bf16
// ---------------------------------------------------------------------------
__global__ __launch_bounds__(256) void rmsnorm_bf16(
    const float* __restrict__ src, int srcStride, int cols,
    const float* __restrict__ g,
    unsigned short* __restrict__ dst, int dstStride) {

    const int row = blockIdx.x;
    const float* x = src + (size_t)row * srcStride;
    float ss = 0.0f;
    for (int c = threadIdx.x; c < cols; c += blockDim.x) {
        float v = x[c];
        ss += v * v;
    }
    __shared__ float red[8];
#pragma unroll
    for (int off = 16; off > 0; off >>= 1) ss += __shfl_xor(ss, off, 32);
    if ((threadIdx.x & 31) == 0) red[threadIdx.x >> 5] = ss;
    __syncthreads();
    if (threadIdx.x == 0) {
        float t = 0.0f;
#pragma unroll
        for (int i = 0; i < 8; ++i) t += red[i];
        red[0] = t;
    }
    __syncthreads();
    float norm = sqrtf(red[0]) * rsqrtf((float)cols);
    float inv  = 1.0f / (norm + 1e-6f);
    for (int c = threadIdx.x; c < cols; c += blockDim.x)
        dst[(size_t)row * dstStride + c] = f32_to_bf16u(x[c] * inv * g[c]);
}

// ---------------------------------------------------------------------------
// RoPE helpers + layout assembly into [b][h][s][d] bf16 tensors
// ---------------------------------------------------------------------------
__device__ __forceinline__ float rope_val(const float* base, int j, int s) {
    int j2 = j & 31;
    float invf = __expf(-9.210340372f * (float)(2 * j2) / 64.0f); // 10000^{-2j2/64}
    float ang = (float)s * invf;
    float c = __cosf(ang), sn = __sinf(ang);
    float x = base[j];
    float other = (j < 32) ? base[j + 32] : base[j - 32];
    return (j < 32) ? (x * c - other * sn) : (x * c + other * sn);
}

__global__ void build_query(const float* __restrict__ q,   // (2048 x 6144) f32
                            unsigned short* __restrict__ query) { // [b][h][s][192]
    size_t i = (size_t)blockIdx.x * blockDim.x + threadIdx.x;
    size_t total = (size_t)C_B * C_NH * C_S * C_QK;
    if (i >= total) return;
    int d = (int)(i % C_QK); size_t t = i / C_QK;
    int s = (int)(t % C_S);  t /= C_S;
    int h = (int)(t % C_NH);
    int b = (int)(t / C_NH);
    const float* qp = q + ((size_t)b * C_S + s) * (C_NH * C_QK) + h * C_QK;
    float val = (d < C_ND) ? qp[d] : rope_val(qp + C_ND, d - C_ND, s);
    query[i] = f32_to_bf16u(val);
}

__global__ void build_key(const float* __restrict__ kst,  // (2048 x 8192) f32
                          const float* __restrict__ kv,   // (2048 x KVPAD) f32
                          unsigned short* __restrict__ key) { // [b][h][s][192]
    size_t i = (size_t)blockIdx.x * blockDim.x + threadIdx.x;
    size_t total = (size_t)C_B * C_NH * C_S * C_QK;
    if (i >= total) return;
    int d = (int)(i % C_QK); size_t t = i / C_QK;
    int s = (int)(t % C_S);  t /= C_S;
    int h = (int)(t % C_NH);
    int b = (int)(t / C_NH);
    size_t row = (size_t)b * C_S + s;
    float val;
    if (d < C_ND) val = kst[row * (C_NH * 256) + h * 256 + d];
    else          val = rope_val(kv + row * KVPAD + C_KVR, d - C_ND, s);
    key[i] = f32_to_bf16u(val);
}

__global__ void build_v(const float* __restrict__ kst,
                        unsigned short* __restrict__ v) {  // [b][h][s][128]
    size_t i = (size_t)blockIdx.x * blockDim.x + threadIdx.x;
    size_t total = (size_t)C_B * C_NH * C_S * C_VD;
    if (i >= total) return;
    int d = (int)(i % C_VD); size_t t = i / C_VD;
    int s = (int)(t % C_S);  t /= C_S;
    int h = (int)(t % C_NH);
    int b = (int)(t / C_NH);
    size_t row = (size_t)b * C_S + s;
    v[i] = f32_to_bf16u(kst[row * (C_NH * 256) + h * 256 + C_ND + d]);
}

// ---------------------------------------------------------------------------
// Flash attention (non-causal): Br=64 queries per block, Bc=32 keys/iter.
// 4 waves; wave w owns 16 q rows. S = Q K^T via WMMA, online softmax,
// P@V via WMMA with V transposed in LDS. Output bf16 [b][s][h*128+d].
// ---------------------------------------------------------------------------
#define FBR 64
#define FBC 32
#define QSTR 200   // 192 + 8 pad
#define KSTR 200
#define VSTR 40    // 32 + 8 pad (Vt[d][k])
#define PSTR 40    // per-wave P tile stride

__global__ __launch_bounds__(128) void flash_attn(
    const unsigned short* __restrict__ Qg,  // [b][h][s][192]
    const unsigned short* __restrict__ Kg,  // [b][h][s][192]
    const unsigned short* __restrict__ Vg,  // [b][h][s][128]
    unsigned short* __restrict__ Og) {      // [b][s][h*128+d]

    __shared__ alignas(16) unsigned short Qs[FBR * QSTR];
    __shared__ alignas(16) unsigned short Ks[FBC * KSTR];
    __shared__ alignas(16) unsigned short Vt[C_VD * VSTR];
    __shared__ alignas(16) unsigned short Ps[4 * 16 * PSTR];

    const int tid  = threadIdx.x;
    const int wave = tid >> 5;
    const int lane = tid & 31;
    const int half = lane >> 4;
    const int l16  = lane & 15;
    const int qblk = blockIdx.x, head = blockIdx.y, batch = blockIdx.z;
    const size_t bh = (size_t)batch * C_NH + head;

    const unsigned short* Qbase = Qg + (bh * C_S + (size_t)qblk * FBR) * C_QK;
    const unsigned short* Kbase = Kg + bh * C_S * C_QK;
    const unsigned short* Vbase = Vg + bh * C_S * C_VD;

    // ---- stage Q block once
    for (int idx = tid; idx < FBR * (C_QK / 8); idx += 128) {
        int r = idx / 24, c = (idx % 24) * 8;
        *(uint4*)&Qs[r * QSTR + c] = *(const uint4*)(Qbase + (size_t)r * C_QK + c);
    }
    __syncthreads();

    // ---- hoist Q fragments (6 d-chunks of 32)
    Frag16 qf[6];
    {
        int m = wave * 16 + l16;
#pragma unroll
        for (int kd = 0; kd < 6; ++kd) {
            qf[kd].u[0] = *(const uint4*)&Qs[m * QSTR + kd * 32 + half * 8];
            qf[kd].u[1] = *(const uint4*)&Qs[m * QSTR + kd * 32 + 16 + half * 8];
        }
    }

    const v8f vzero = {0.f, 0.f, 0.f, 0.f, 0.f, 0.f, 0.f, 0.f};
    float mrow[8], lrow[8];
    v8f oacc[8];
#pragma unroll
    for (int r = 0; r < 8; ++r) { mrow[r] = -INFINITY; lrow[r] = 0.0f; }
#pragma unroll
    for (int n = 0; n < 8; ++n) oacc[n] = vzero;

    const float scale = rsqrtf((float)C_QK);
    const int nkb = C_S / FBC;

    for (int kb = 0; kb < nkb; ++kb) {
        __syncthreads();
        // ---- stage K block (32 x 192)
        for (int idx = tid; idx < FBC * 24; idx += 128) {
            int r = idx / 24, c = (idx % 24) * 8;
            *(uint4*)&Ks[r * KSTR + c] =
                *(const uint4*)(Kbase + (size_t)(kb * FBC + r) * C_QK + c);
        }
        // ---- stage V block transposed: Vt[d][k]
        for (int idx = tid; idx < FBC * 16; idx += 128) {
            int r = idx / 16, c = (idx % 16) * 8;
            uint4 p = *(const uint4*)(Vbase + (size_t)(kb * FBC + r) * C_VD + c);
            unsigned short tmp[8];
            *(uint4*)tmp = p;
#pragma unroll
            for (int i = 0; i < 8; ++i) Vt[(c + i) * VSTR + r] = tmp[i];
        }
        __syncthreads();

        // ---- S = Q K^T (two 16-col tiles)
        v8f sacc[2];
#pragma unroll
        for (int nt = 0; nt < 2; ++nt) {
            sacc[nt] = vzero;
            int kc = nt * 16 + l16;   // key column (B operand N index)
#pragma unroll
            for (int kd = 0; kd < 6; ++kd) {
                Frag16 bf;
                bf.u[0] = *(const uint4*)&Ks[kc * KSTR + kd * 32 + half * 8];
                bf.u[1] = *(const uint4*)&Ks[kc * KSTR + kd * 32 + 16 + half * 8];
                sacc[nt] = __builtin_amdgcn_wmma_f32_16x16x32_bf16(
                    false, qf[kd].v, false, bf.v, (short)0, sacc[nt], false, false);
            }
        }

        // ---- online softmax (row = wave*16 + half*8 + r, cols across lanes)
        float fac[8], lad[8];
#pragma unroll
        for (int r = 0; r < 8; ++r) {
            float v = fmaxf(sacc[0][r], sacc[1][r]) * scale;
#pragma unroll
            for (int off = 1; off < 16; off <<= 1)
                v = fmaxf(v, __shfl_xor(v, off, 16));
            float mn = fmaxf(mrow[r], v);
            fac[r] = __expf(mrow[r] - mn);
            mrow[r] = mn;
            lad[r] = 0.0f;
        }
#pragma unroll
        for (int nt = 0; nt < 2; ++nt)
#pragma unroll
            for (int r = 0; r < 8; ++r) {
                float p = __expf(sacc[nt][r] * scale - mrow[r]);
                lad[r] += p;
                Ps[(wave * 16 + half * 8 + r) * PSTR + nt * 16 + l16] = f32_to_bf16u(p);
            }
#pragma unroll
        for (int r = 0; r < 8; ++r) {
            float s = lad[r];
#pragma unroll
            for (int off = 1; off < 16; off <<= 1) s += __shfl_xor(s, off, 16);
            lrow[r] = lrow[r] * fac[r] + s;
        }
#pragma unroll
        for (int n = 0; n < 8; ++n)
#pragma unroll
            for (int r = 0; r < 8; ++r) oacc[n][r] *= fac[r];

        // ---- O += P @ V  (k = 32 -> single WMMA k-step per n-tile)
        Frag16 pf;
        {
            int m = wave * 16 + l16;
            pf.u[0] = *(const uint4*)&Ps[m * PSTR + half * 8];
            pf.u[1] = *(const uint4*)&Ps[m * PSTR + 16 + half * 8];
        }
#pragma unroll
        for (int n = 0; n < 8; ++n) {
            Frag16 vf;
            int d = n * 16 + l16;
            vf.u[0] = *(const uint4*)&Vt[d * VSTR + half * 8];
            vf.u[1] = *(const uint4*)&Vt[d * VSTR + 16 + half * 8];
            oacc[n] = __builtin_amdgcn_wmma_f32_16x16x32_bf16(
                false, pf.v, false, vf.v, (short)0, oacc[n], false, false);
        }
    }

    // ---- epilogue: O /= l, write bf16 to [b][s][h*128+d]
#pragma unroll
    for (int n = 0; n < 8; ++n)
#pragma unroll
        for (int r = 0; r < 8; ++r) {
            int q = qblk * FBR + wave * 16 + half * 8 + r;
            float o = oacc[n][r] / lrow[r];
            size_t oi = ((size_t)batch * C_S + q) * (C_NH * C_VD)
                      + (size_t)head * C_VD + n * 16 + l16;
            Og[oi] = f32_to_bf16u(o);
        }
}

// ---------------------------------------------------------------------------
// Host-side orchestration
// ---------------------------------------------------------------------------
extern "C" void kernel_launch(void* const* d_in, const int* in_sizes, int n_in,
                              void* d_out, int out_size, void* d_ws, size_t ws_size,
                              hipStream_t stream) {
    (void)in_sizes; (void)n_in; (void)out_size; (void)ws_size;

    const float* hs    = (const float*)d_in[0];   // (2,1024,4096)
    const float* w_qa  = (const float*)d_in[1];   // (4096,1536)
    const float* g_qa  = (const float*)d_in[2];   // (1536,)
    const float* w_qb  = (const float*)d_in[3];   // (1536,6144)
    const float* w_kva = (const float*)d_in[4];   // (4096,576)
    const float* g_kva = (const float*)d_in[5];   // (512,)
    const float* w_kvb = (const float*)d_in[6];   // (512,8192)
    const float* w_o   = (const float*)d_in[7];   // (4096,4096)
    float* out = (float*)d_out;                   // (2,1024,4096)

    char* p = (char*)d_ws;
    auto alloc = [&](size_t bytes) -> void* {
        void* r = (void*)p;
        p += (bytes + 255) & ~(size_t)255;
        return r;
    };

    unsigned short* hs_bf   = (unsigned short*)alloc((size_t)C_M * C_H * 2);
    unsigned short* wqa_bf  = (unsigned short*)alloc((size_t)C_H * C_QR * 2);
    unsigned short* wqb_bf  = (unsigned short*)alloc((size_t)C_QR * C_NH * C_QK * 2);
    unsigned short* wkva_bf = (unsigned short*)alloc((size_t)C_H * KVPAD * 2);  // padded 576->768
    unsigned short* wkvb_bf = (unsigned short*)alloc((size_t)C_KVR * C_NH * 256 * 2);
    unsigned short* wo_bf   = (unsigned short*)alloc((size_t)C_H * C_H * 2);
    float*          qa_f    = (float*)alloc((size_t)C_M * C_QR * 4);
    unsigned short* qc_bf   = (unsigned short*)alloc((size_t)C_M * C_QR * 2);
    float*          q_f     = (float*)alloc((size_t)C_M * C_NH * C_QK * 4);
    float*          kv_f    = (float*)alloc((size_t)C_M * KVPAD * 4);
    unsigned short* kc_bf   = (unsigned short*)alloc((size_t)C_M * C_KVR * 2);
    float*          kst_f   = (float*)alloc((size_t)C_M * C_NH * 256 * 4);
    unsigned short* query   = (unsigned short*)alloc((size_t)C_B * C_NH * C_S * C_QK * 2);
    unsigned short* keybf   = (unsigned short*)alloc((size_t)C_B * C_NH * C_S * C_QK * 2);
    unsigned short* vbf     = (unsigned short*)alloc((size_t)C_B * C_NH * C_S * C_VD * 2);
    unsigned short* attn_bf = (unsigned short*)alloc((size_t)C_M * C_H * 2);

    auto cvt = [&](const float* s, unsigned short* d, int rows, int cols, int dcols) {
        size_t total = (size_t)rows * dcols;
        cvt_f32_bf16_pad<<<(unsigned)((total + 255) / 256), 256, 0, stream>>>(s, d, rows, cols, dcols);
    };

    // 1) downconvert everything to bf16 (weights once, activations once)
    cvt(hs,    hs_bf,   C_M,   C_H,          C_H);
    cvt(w_qa,  wqa_bf,  C_H,   C_QR,         C_QR);
    cvt(w_qb,  wqb_bf,  C_QR,  C_NH * C_QK,  C_NH * C_QK);
    cvt(w_kva, wkva_bf, C_H,   C_KVR + C_RD, KVPAD);
    cvt(w_kvb, wkvb_bf, C_KVR, C_NH * 256,   C_NH * 256);
    cvt(w_o,   wo_bf,   C_H,   C_H,          C_H);

    auto gemm = [&](const unsigned short* A, const unsigned short* B, float* C,
                    int M, int N, int K, int lda, int ldb, int ldc) {
        dim3 grid(N / GBN, M / GBM);
        gemm_bf16_wmma<<<grid, 256, 0, stream>>>(A, B, C, K, lda, ldb, ldc);
    };

    // 2) q_a = hs @ w_qa ; q_c = rmsnorm(q_a, g_qa)
    gemm(hs_bf, wqa_bf, qa_f, C_M, C_QR, C_H, C_H, C_QR, C_QR);
    rmsnorm_bf16<<<C_M, 256, 0, stream>>>(qa_f, C_QR, C_QR, g_qa, qc_bf, C_QR);

    // 3) q = q_c @ w_qb
    gemm(qc_bf, wqb_bf, q_f, C_M, C_NH * C_QK, C_QR, C_QR, C_NH * C_QK, C_NH * C_QK);

    // 4) kv = hs @ w_kva (padded N=768); k_c = rmsnorm(kv[:,:512], g_kva)
    gemm(hs_bf, wkva_bf, kv_f, C_M, KVPAD, C_H, C_H, KVPAD, KVPAD);
    rmsnorm_bf16<<<C_M, 256, 0, stream>>>(kv_f, KVPAD, C_KVR, g_kva, kc_bf, C_KVR);

    // 5) kst = k_c @ w_kvb
    gemm(kc_bf, wkvb_bf, kst_f, C_M, C_NH * 256, C_KVR, C_KVR, C_NH * 256, C_NH * 256);

    // 6) RoPE + assemble query/key/v in [b][h][s][d] bf16
    {
        size_t tq = (size_t)C_B * C_NH * C_S * C_QK;
        build_query<<<(unsigned)((tq + 255) / 256), 256, 0, stream>>>(q_f, query);
        build_key<<<(unsigned)((tq + 255) / 256), 256, 0, stream>>>(kst_f, kv_f, keybf);
        size_t tv = (size_t)C_B * C_NH * C_S * C_VD;
        build_v<<<(unsigned)((tv + 255) / 256), 256, 0, stream>>>(kst_f, vbf);
    }

    // 7) flash attention -> attn_bf [b][s][h*128+d]
    {
        dim3 grid(C_S / FBR, C_NH, C_B);
        flash_attn<<<grid, 128, 0, stream>>>(query, keybf, vbf, attn_bf);
    }

    // 8) out = attn @ w_o  (f32 straight into d_out)
    gemm(attn_bf, wo_bf, out, C_M, C_H, C_H, C_H, C_H, C_H);
}
